// MultiEnvironmentPredictor_39642548142647
// MI455X (gfx1250) — compile-verified, matrix operand order checked by reading, not run on hardware
//
#include <hip/hip_runtime.h>

// Problem dimensions (fixed by the reference)
#define BB  16384
#define II  512
#define EE  8
#define INVD 128
#define SSD 64
#define HHD 256
#define H2D 128

#define AS1 __attribute__((address_space(1)))
#define AS3 __attribute__((address_space(3)))

typedef __attribute__((ext_vector_type(8)))  __bf16 v8bf;
typedef __attribute__((ext_vector_type(16))) __bf16 v16bf;
typedef __attribute__((ext_vector_type(8)))  float  v8f;
typedef int v4i __attribute__((vector_size(16)));   // matches builtin's pointee type

__device__ __forceinline__ unsigned short f2bfbits(float f) {
  union { float f; unsigned u; } v; v.f = f;
  unsigned r = v.u + 0x7FFFu + ((v.u >> 16) & 1u);   // round-to-nearest-even
  return (unsigned short)(r >> 16);
}
__device__ __forceinline__ float bfbits2f(unsigned short h) {
  union { unsigned u; float f; } v; v.u = ((unsigned)h) << 16; return v.f;
}

// ---------------------------------------------------------------------------
// f32 -> bf16 flat convert (4 elements / thread)
__global__ void k_convert_bf16(const float* __restrict__ src,
                               unsigned short* __restrict__ dst, int n4) {
  int t = blockIdx.x * blockDim.x + threadIdx.x;
  if (t >= n4) return;
  float4 v = ((const float4*)src)[t];
  ushort4 o;
  o.x = f2bfbits(v.x); o.y = f2bfbits(v.y); o.z = f2bfbits(v.z); o.w = f2bfbits(v.w);
  ((ushort4*)dst)[t] = o;
}

// f32 [batch,K,N] -> bf16 [batch,N,K]  (so B-fragment columns are K-contiguous)
__global__ void k_transpose_bf16(const float* __restrict__ src,
                                 unsigned short* __restrict__ dst,
                                 int batch, int K, int N) {
  int t = blockIdx.x * blockDim.x + threadIdx.x;
  int total = batch * K * N;
  if (t >= total) return;
  int k = t % K;
  int n = (t / K) % N;
  int b = t / (K * N);
  dst[t] = f2bfbits(src[((size_t)b * K + k) * N + n]);
}

// ---------------------------------------------------------------------------
// 16-element bf16 fragment per lane (CDNA5 16-bit A/B layout):
// lane<16 holds K = base+{0..7,16..23}, lane>=16 holds K = base+{8..15,24..31};
// with `p` pre-offset by (lane>>4)*8 this is two contiguous 16-byte loads.
__device__ __forceinline__ v16bf ld_frag(const unsigned short* p) {
  v8bf a = *(const v8bf*)(const void*)p;
  v8bf b = *(const v8bf*)(const void*)(p + 16);
  return __builtin_shufflevector(a, b, 0,1,2,3,4,5,6,7,8,9,10,11,12,13,14,15);
}

// Async copy of one 16-byte chunk global -> LDS (CDNA5 ASYNCcnt path)
__device__ __forceinline__ void async_copy16(const unsigned short* g,
                                             unsigned short* l_generic) {
#if __has_builtin(__builtin_amdgcn_global_load_async_to_lds_b128)
  __builtin_amdgcn_global_load_async_to_lds_b128(
      (AS1 v4i*)g, (AS3 v4i*)l_generic, 0, 0);
#else
  unsigned lds_off = (unsigned)(uintptr_t)l_generic;   // fallback (unused on ROCm 7.2)
  unsigned long long ga = (unsigned long long)(uintptr_t)g;
  asm volatile("global_load_async_to_lds_b128 %0, %1, off"
               :: "v"(lds_off), "v"(ga) : "memory");
#endif
}

// ---------------------------------------------------------------------------
// GEMM: out[M,N] = act(A[M,K](bf16) @ Bt[e][N,K](bf16)^T + bias[e][N])
// Block = (32,8): 8 waves, 128 rows x 64 cols. The 64xK B panel is staged
// once per block into LDS via async-to-LDS, then all 8 waves feed WMMA B
// fragments from LDS while A fragments are double-buffered from global.
// env != null: only rows with env[row] == blockIdx.z are stored.
__global__ void __launch_bounds__(256)
k_gemm_wmma(const unsigned short* __restrict__ A,
            const unsigned short* __restrict__ Bt,
            const float* __restrict__ bias,
            float* __restrict__ outF, unsigned short* __restrict__ outB,
            const int* __restrict__ env,
            int M, int N, int K, int relu) {
  extern __shared__ unsigned short sB[];   // [64][K] bf16 panel

  const int lane = threadIdx.x;
  const int l15  = lane & 15;
  const int half = lane >> 4;
  const int e    = blockIdx.z;
  const int tn   = blockIdx.x * 64;
  const int tm   = (blockIdx.y * blockDim.y + threadIdx.y) * 16;

  const unsigned short* Be = Bt + (size_t)e * N * K;
  const float*          be = bias + (size_t)e * N;

  // --- cooperative async staging of the contiguous 64*K panel ---
  {
    const unsigned short* gpanel = Be + (size_t)tn * K;
    const int tid     = threadIdx.y * 32 + lane;
    const int nchunks = (64 * K) / 8;            // 16-byte chunks
    for (int c = tid; c < nchunks; c += 256) {
      async_copy16(gpanel + c * 8, sB + c * 8);
    }
    asm volatile("s_wait_asynccnt 0x0" ::: "memory");
    __syncthreads();
  }

  if (env) {  // wave-uniform exit if no row of this 16-row tile matches env e
    bool rm = (env[tm + l15] == e);
    if (__ballot((int)rm) == 0ull) return;
  }

  const unsigned short* ap = A + (size_t)(tm + l15) * K + half * 8;
  const unsigned short* bp = sB + (size_t)l15 * K + half * 8;   // + t*16*K per tile

  v8f acc[4] = {};
  v16bf a_cur = ld_frag(ap);
  for (int k = 0; k < K; k += 32) {
    v16bf a_nxt = a_cur;
    if (k + 32 < K) a_nxt = ld_frag(ap + k + 32);   // prefetch under the WMMAs
#pragma unroll
    for (int t = 0; t < 4; ++t) {
      v16bf bv = ld_frag(bp + (size_t)t * 16 * K + k);  // LDS (ds_load_b128)
      acc[t] = __builtin_amdgcn_wmma_f32_16x16x32_bf16(
          false, a_cur, false, bv, (short)0, acc[t], false, false);
    }
    a_cur = a_nxt;
  }

#pragma unroll
  for (int t = 0; t < 4; ++t) {
    float bn = be[tn + t * 16 + l15];
#pragma unroll
    for (int i = 0; i < 8; ++i) {
      int row = tm + half * 8 + i;            // C/D layout: VGPR i -> M = i + 8*half
      float v = acc[t][i] + bn;
      if (relu) v = fmaxf(v, 0.0f);
      if (!env || env[row] == e) {
        size_t idx = (size_t)row * N + tn + t * 16 + l15;
        if (outF) outF[idx] = v;
        if (outB) outB[idx] = f2bfbits(v);
      }
    }
  }
}

// ---------------------------------------------------------------------------
// Tail: logits = [inv|spec] @ Wf + bf ; domain = hd @ Wd2 + bd2  (tiny GEMVs)
__global__ void k_tail(const float* __restrict__ inv, const float* __restrict__ spec,
                       const unsigned short* __restrict__ hd,
                       const float* __restrict__ Wf, const float* __restrict__ bf,
                       const float* __restrict__ Wd2, const float* __restrict__ bd2,
                       float* __restrict__ logits, float* __restrict__ dom) {
  int b = blockIdx.x * blockDim.x + threadIdx.x;
  if (b >= BB) return;
  const float* ib = inv  + (size_t)b * INVD;
  const float* sb = spec + (size_t)b * SSD;
  float acc = bf[0];
  for (int i = 0; i < INVD; ++i) acc += ib[i] * Wf[i];
  for (int j = 0; j < SSD;  ++j) acc += sb[j] * Wf[INVD + j];
  logits[b] = acc;

  float d[EE];
#pragma unroll
  for (int e2 = 0; e2 < EE; ++e2) d[e2] = bd2[e2];
  const unsigned short* hb = hd + (size_t)b * H2D;
  for (int i = 0; i < H2D; ++i) {
    float h = bfbits2f(hb[i]);
#pragma unroll
    for (int e2 = 0; e2 < EE; ++e2) d[e2] += h * Wd2[i * EE + e2];
  }
#pragma unroll
  for (int e2 = 0; e2 < EE; ++e2) dom[(size_t)b * EE + e2] = d[e2];
}

// ---------------------------------------------------------------------------
extern "C" void kernel_launch(void* const* d_in, const int* in_sizes, int n_in,
                              void* d_out, int out_size, void* d_ws, size_t ws_size,
                              hipStream_t stream) {
  (void)in_sizes; (void)n_in; (void)out_size; (void)ws_size;
  const float* x   = (const float*)d_in[0];
  const int*   env = (const int*)  d_in[1];
  const float* W1  = (const float*)d_in[2];
  const float* b1  = (const float*)d_in[3];
  const float* W2  = (const float*)d_in[4];
  const float* b2  = (const float*)d_in[5];
  const float* Ws1 = (const float*)d_in[6];
  const float* bs1 = (const float*)d_in[7];
  const float* Ws2 = (const float*)d_in[8];
  const float* bs2 = (const float*)d_in[9];
  const float* Wf  = (const float*)d_in[10];
  const float* bf  = (const float*)d_in[11];
  const float* Wd1 = (const float*)d_in[12];
  const float* bd1 = (const float*)d_in[13];
  const float* Wd2 = (const float*)d_in[14];
  const float* bd2 = (const float*)d_in[15];

  // Output layout: logits[B,1] | invariant[B,128] | specific[B,64] | domain[B,8]
  float* out    = (float*)d_out;
  float* logits = out;
  float* invF   = out + BB;
  float* specF  = invF + (size_t)BB * INVD;
  float* domF   = specF + (size_t)BB * SSD;

  // Workspace carve (256-B aligned), ~39 MB total
  char* w = (char*)d_ws;
  auto carve = [&](size_t bytes) {
    char* p = w; w += (bytes + 255) & ~(size_t)255; return p;
  };
  unsigned short* xb   = (unsigned short*)carve((size_t)BB * II  * 2);
  unsigned short* h1   = (unsigned short*)carve((size_t)BB * HHD * 2);
  unsigned short* invB = (unsigned short*)carve((size_t)BB * INVD* 2);
  unsigned short* hs   = (unsigned short*)carve((size_t)BB * H2D * 2);
  unsigned short* hd   = (unsigned short*)carve((size_t)BB * H2D * 2);
  unsigned short* W1t  = (unsigned short*)carve((size_t)HHD * II  * 2);
  unsigned short* W2t  = (unsigned short*)carve((size_t)INVD* HHD * 2);
  unsigned short* Ws1t = (unsigned short*)carve((size_t)EE * H2D * II * 2);
  unsigned short* Ws2t = (unsigned short*)carve((size_t)EE * SSD * H2D * 2);
  unsigned short* Wd1t = (unsigned short*)carve((size_t)H2D * INVD * 2);

  // 1) Convert x to bf16
  { int n4 = BB * II / 4;
    k_convert_bf16<<<(n4 + 255) / 256, 256, 0, stream>>>(x, xb, n4); }

  // 2) Convert+transpose weights to bf16 [N,K]
  auto tr = [&](const float* s, unsigned short* d, int batch, int K, int N) {
    int tot = batch * K * N;
    k_transpose_bf16<<<(tot + 255) / 256, 256, 0, stream>>>(s, d, batch, K, N);
  };
  tr(W1,  W1t,  1,  II,  HHD);
  tr(W2,  W2t,  1,  HHD, INVD);
  tr(Ws1, Ws1t, EE, II,  H2D);
  tr(Ws2, Ws2t, EE, H2D, SSD);
  tr(Wd1, Wd1t, 1,  INVD, H2D);

  dim3 blk(32, 8, 1);  // 8 waves/block: 128 rows x 64 cols, shared LDS B panel
  auto lds = [](int K) { return (size_t)64 * K * 2; };
  // h1 = relu(x @ W1 + b1)                         [B,256]
  k_gemm_wmma<<<dim3(HHD/64,  BB/128, 1),  blk, lds(II),   stream>>>(xb,  W1t,  b1,  nullptr, h1,   nullptr, BB, HHD,  II,  1);
  // invariant = h1 @ W2 + b2                       [B,128] (f32 to d_out + bf16 copy)
  k_gemm_wmma<<<dim3(INVD/64, BB/128, 1),  blk, lds(HHD),  stream>>>(h1,  W2t,  b2,  invF,    invB, nullptr, BB, INVD, HHD, 0);
  // hs = relu(x @ Ws1[e] + bs1[e]) where env==e    [B,128]
  k_gemm_wmma<<<dim3(H2D/64,  BB/128, EE), blk, lds(II),   stream>>>(xb,  Ws1t, bs1, nullptr, hs,   env,     BB, H2D,  II,  1);
  // specific = hs @ Ws2[e] + bs2[e] where env==e   [B,64] (f32 to d_out)
  k_gemm_wmma<<<dim3(SSD/64,  BB/128, EE), blk, lds(H2D),  stream>>>(hs,  Ws2t, bs2, specF,   nullptr, env,   BB, SSD,  H2D, 0);
  // hd = relu(invariant @ Wd1 + bd1)               [B,128]
  k_gemm_wmma<<<dim3(H2D/64,  BB/128, 1),  blk, lds(INVD), stream>>>(invB, Wd1t, bd1, nullptr, hd,   nullptr, BB, H2D,  INVD, 1);

  // 3) Tail: logits + domain_logits
  k_tail<<<(BB + 255) / 256, 256, 0, stream>>>(invF, specF, hd, Wf, bf, Wd2, bd2, logits, domF);
}
